// EvoformerODEFunc_9543417331857
// MI455X (gfx1250) — matrix operand order repaired and checked by reading, not compiled, not run on hardware
//
#include <hip/hip_runtime.h>
#include <cstddef>

// ---------------------------------------------------------------------------
// EvoformerODEFunc for MI455X (gfx1250, wave32, WMMA).
// Strategy: every block is row-wise LN -> GEMM(-ReLU-GEMM); both swapaxes
// wrappers are mathematical no-ops. All GEMMs run through one bf16 WMMA
// kernel (v_wmma_f32_16x16x32_bf16, f32 accum). Outer-product-mean is a
// per-(i,j) 32x32x128 WMMA GEMM, chunked so the op buffer is 16MB.
// Assumptions: inputs flattened in setup_inputs() dict insertion order;
// ws_size >= ~224MB.
// ---------------------------------------------------------------------------

typedef __bf16 bf16;
typedef __attribute__((ext_vector_type(16))) __bf16 v16bf;
typedef __attribute__((ext_vector_type(8)))  __bf16 v8bf;
typedef __attribute__((ext_vector_type(8)))  float  v8f;

#define WMMA_BF16(a, b, c) \
  __builtin_amdgcn_wmma_f32_16x16x32_bf16(false, (a), false, (b), (short)0, (c), false, false)

// ---- fragment loaders (CDNA5 ISA 7.12.2 layouts) --------------------------
// A 16x32 bf16: lane L -> M row (L&15); half=L>>4 selects K interleave:
//   elements 0..7  -> K = half*8 + 0..7        (16B contiguous)
//   elements 8..15 -> K = 16 + half*8 + 0..7   (16B contiguous)
__device__ inline v16bf load_a_frag(const bf16* __restrict__ A, int lda,
                                    int m0, int k0, int lane) {
  const int row  = m0 + (lane & 15);
  const int half = lane >> 4;
  const bf16* p0 = A + (size_t)row * lda + k0 + half * 8;
  v8bf lo = *(const v8bf*)p0;
  v8bf hi = *(const v8bf*)(p0 + 16);
  return __builtin_shufflevector(lo, hi, 0, 1, 2, 3, 4, 5, 6, 7,
                                 8, 9, 10, 11, 12, 13, 14, 15);
}

// B 32x16 bf16: lane L -> N col (L&15); half=L>>4 selects K like A:
//   elements 0..7 -> K = half*8 + 0..7, elements 8..15 -> K = 16 + half*8 + 0..7
__device__ inline v16bf load_b_frag(const bf16* __restrict__ B, long ldb,
                                    int k0, int n0, int lane) {
  const int col  = n0 + (lane & 15);
  const int half = lane >> 4;
  v16bf r;
#pragma unroll
  for (int s = 0; s < 8; ++s) {
    r[s]     = B[(size_t)(k0 + half * 8 + s) * ldb + col];
    r[s + 8] = B[(size_t)(k0 + 16 + half * 8 + s) * ldb + col];
  }
  return r;
}

// ---------------------------------------------------------------------------
// Generic bf16 GEMM: C[M,Nout] = op(A[M,K] @ B[K,N] + bias)(+res)
// BM=128, BN=64; 8 waves in 4(M)x2(N); each wave owns 32x32 (4 WMMA tiles).
// M % 128 == 0, N % 64 == 0, K % 32 == 0. Nout <= N handles zero-padded B.
// ---------------------------------------------------------------------------
__global__ __launch_bounds__(256) void gemm_bf16_kernel(
    const bf16* __restrict__ A, const bf16* __restrict__ B,
    const float* __restrict__ bias, const float* __restrict__ res,
    float* __restrict__ Cf, bf16* __restrict__ Cb,
    int M, int N, int K, int Nout, int relu) {
  const int lane  = threadIdx.x & 31;
  const int wave  = threadIdx.x >> 5;
  const int m_blk = blockIdx.y * 128 + (wave >> 1) * 32;
  const int n_blk = blockIdx.x * 64 + (wave & 1) * 32;

  v8f acc[2][2] = {};
  for (int k0 = 0; k0 < K; k0 += 32) {
    v16bf a0 = load_a_frag(A, K, m_blk,      k0, lane);
    v16bf a1 = load_a_frag(A, K, m_blk + 16, k0, lane);
    v16bf b0 = load_b_frag(B, N, k0, n_blk,      lane);
    v16bf b1 = load_b_frag(B, N, k0, n_blk + 16, lane);
    acc[0][0] = WMMA_BF16(a0, b0, acc[0][0]);
    acc[0][1] = WMMA_BF16(a0, b1, acc[0][1]);
    acc[1][0] = WMMA_BF16(a1, b0, acc[1][0]);
    acc[1][1] = WMMA_BF16(a1, b1, acc[1][1]);
  }

  // C/D layout: lane L -> col L&15, rows 8*(L>>4)+r (r = accum element)
  const int ct = lane & 15;
  const int rh = (lane >> 4) * 8;
#pragma unroll
  for (int mt = 0; mt < 2; ++mt) {
#pragma unroll
    for (int nt = 0; nt < 2; ++nt) {
      const int col = n_blk + nt * 16 + ct;
      if (col >= Nout) continue;
      const float bv = bias ? bias[col] : 0.0f;
#pragma unroll
      for (int r = 0; r < 8; ++r) {
        const int row = m_blk + mt * 16 + rh + r;
        float y = acc[mt][nt][r] + bv;
        if (relu) y = fmaxf(y, 0.0f);
        const size_t idx = (size_t)row * Nout + col;
        if (res) y += res[idx];
        if (Cf) Cf[idx] = y;
        else    Cb[idx] = (bf16)y;
      }
    }
  }
  (void)M;
}

// ---------------------------------------------------------------------------
// Outer product mean: op[i,j,h,k] = sum_s a[s,i,h]*b[s,j,k]/S  (32x32, K=128)
// At: [256,32,128] bf16 (i,h,s) pre-scaled by 1/128;  Bt: [128,256,32] (s,j,k)
// One wave per (i,j); block = 8 j's; grid = (32 j-tiles, 32 i's per chunk).
// ---------------------------------------------------------------------------
__global__ __launch_bounds__(256) void outer_kernel(
    const bf16* __restrict__ At, const bf16* __restrict__ Bt,
    bf16* __restrict__ Op, int i0) {
  const int lane = threadIdx.x & 31;
  const int wave = threadIdx.x >> 5;
  const int j    = blockIdx.x * 8 + wave;
  const int ii   = blockIdx.y;
  const bf16* Ai = At + (size_t)(i0 + ii) * 32 * 128;
  const bf16* Bj = Bt + (size_t)j * 32;  // row stride 256*32

  v8f acc[2][2] = {};
#pragma unroll
  for (int k0 = 0; k0 < 128; k0 += 32) {
    v16bf a0 = load_a_frag(Ai, 128, 0,  k0, lane);
    v16bf a1 = load_a_frag(Ai, 128, 16, k0, lane);
    v16bf b0 = load_b_frag(Bj, 256 * 32, k0, 0,  lane);
    v16bf b1 = load_b_frag(Bj, 256 * 32, k0, 16, lane);
    acc[0][0] = WMMA_BF16(a0, b0, acc[0][0]);
    acc[0][1] = WMMA_BF16(a0, b1, acc[0][1]);
    acc[1][0] = WMMA_BF16(a1, b0, acc[1][0]);
    acc[1][1] = WMMA_BF16(a1, b1, acc[1][1]);
  }

  bf16* out = Op + ((size_t)ii * 256 + j) * 1024;  // row-major h*32+k flatten
  const int ct = lane & 15;
  const int rh = (lane >> 4) * 8;
#pragma unroll
  for (int mt = 0; mt < 2; ++mt)
#pragma unroll
    for (int nt = 0; nt < 2; ++nt)
#pragma unroll
      for (int r = 0; r < 8; ++r)
        out[(mt * 16 + rh + r) * 32 + nt * 16 + ct] = (bf16)acc[mt][nt][r];
}

// ---------------------------------------------------------------------------
// LayerNorm (per row over last dim d) -> bf16. One wave32 per row.
// ---------------------------------------------------------------------------
__global__ __launch_bounds__(256) void ln_bf16_kernel(
    const float* __restrict__ X, const float* __restrict__ g,
    const float* __restrict__ b, bf16* __restrict__ Y, int rows, int d) {
  const int lane = threadIdx.x & 31;
  const long row = (long)blockIdx.x * 8 + (threadIdx.x >> 5);
  if (row >= rows) return;
  const float* x = X + row * (long)d;
  float s = 0.0f, ss = 0.0f;
  for (int i = lane; i < d; i += 32) { float v = x[i]; s += v; ss += v * v; }
#pragma unroll
  for (int off = 16; off; off >>= 1) {
    s  += __shfl_xor(s,  off, 32);
    ss += __shfl_xor(ss, off, 32);
  }
  const float mu  = s / d;
  const float var = ss / d - mu * mu;
  const float inv = rsqrtf(var + 1e-5f);
  bf16* y = Y + row * (long)d;
  for (int i = lane; i < d; i += 32)
    y[i] = (bf16)((x[i] - mu) * inv * g[i] + b[i]);
}

// ---- small elementwise helpers --------------------------------------------
__global__ void convert_bf16_kernel(const float* __restrict__ src,
                                    bf16* __restrict__ dst, long n) {
  long i = (long)blockIdx.x * blockDim.x + threadIdx.x;
  if (i < n) dst[i] = (bf16)src[i];
}

__global__ void pad_weight_kernel(const float* __restrict__ src,
                                  bf16* __restrict__ dst, int K, int Nsrc,
                                  int Npad) {
  int i = blockIdx.x * blockDim.x + threadIdx.x;
  if (i >= K * Npad) return;
  int k = i / Npad, n = i % Npad;
  dst[i] = (bf16)(n < Nsrc ? src[k * Nsrc + n] : 0.0f);
}

// a_f32 [S=128, R=256, 32] -> At bf16 [256, 32, 128] scaled by 1/S
__global__ void transpose_a_kernel(const float* __restrict__ a,
                                   bf16* __restrict__ at) {
  int i = blockIdx.x * blockDim.x + threadIdx.x;
  if (i >= 256 * 32 * 128) return;
  int s = i & 127, h = (i >> 7) & 31, ir = i >> 12;
  at[i] = (bf16)(a[((size_t)s * 256 + ir) * 32 + h] * (1.0f / 128.0f));
}

// pbm[i,c] = mean_j pb[i,j,c]   (pb: [256,256,256])
__global__ void mean_j_kernel(const float* __restrict__ pb,
                              float* __restrict__ pbm) {
  int i = blockIdx.x, c = threadIdx.x;
  float s = 0.0f;
  for (int j = 0; j < 256; ++j) s += pb[((size_t)(i * 256 + j)) * 256 + c];
  pbm[i * 256 + c] = s * (1.0f / 256.0f);
}

// m_col_in = m + m_row + pbm[r,:] broadcast  (layout [s,r,c], c=256, r=256)
__global__ void mci_kernel(const float* __restrict__ m,
                           const float* __restrict__ mrow,
                           const float* __restrict__ pbm,
                           float* __restrict__ out, long n) {
  long i = (long)blockIdx.x * blockDim.x + threadIdx.x;
  if (i >= n) return;
  int c = (int)(i & 255), r = (int)((i >> 8) & 255);
  out[i] = m[i] + mrow[i] + pbm[r * 256 + c];
}

__global__ void final_kernel(const float* __restrict__ xo,
                             const float* __restrict__ xi,
                             const float* __restrict__ mix, int which,
                             float* __restrict__ dst, long n) {
  long i = (long)blockIdx.x * blockDim.x + threadIdx.x;
  if (i < n) dst[i] = (xo[i] - xi[i]) * mix[which];
}

// time gating MLP: 1->32 SiLU -> 2 -> sigmoid  (one wave)
__global__ void time_gate_kernel(const float* __restrict__ t,
                                 const float* __restrict__ W1,
                                 const float* __restrict__ b1,
                                 const float* __restrict__ W2,
                                 const float* __restrict__ b2,
                                 float* __restrict__ mix) {
  int lane = threadIdx.x;
  float u  = t[0] * W1[lane] + b1[lane];
  float th = u / (1.0f + __expf(-u));  // SiLU
  float p0 = th * W2[lane * 2 + 0];
  float p1 = th * W2[lane * 2 + 1];
#pragma unroll
  for (int off = 16; off; off >>= 1) {
    p0 += __shfl_xor(p0, off, 32);
    p1 += __shfl_xor(p1, off, 32);
  }
  if (lane == 0) {
    mix[0] = 1.0f / (1.0f + __expf(-(p0 + b2[0])));
    mix[1] = 1.0f / (1.0f + __expf(-(p1 + b2[1])));
  }
}

// ---------------------------------------------------------------------------
extern "C" void kernel_launch(void* const* d_in, const int* in_sizes, int n_in,
                              void* d_out, int out_size, void* d_ws,
                              size_t ws_size, hipStream_t stream) {
  (void)in_sizes; (void)n_in; (void)out_size; (void)ws_size;
  const int S = 128, R = 256, CM = 256, CZ = 128;
  const int SR = S * R;      // 32768 rows
  const int RR = R * R;      // 65536 rows

  // ---- inputs (setup_inputs dict insertion order) ----
  int p = 0;
  auto nxt = [&]() { return (const float*)d_in[p++]; };
  const float* t = nxt();
  const float* m = nxt();
  const float* z = nxt();
  // msa_row
  const float *mr_g = nxt(), *mr_bl = nxt(), *mr_W1 = nxt(), *mr_b1 = nxt(),
              *mr_W2 = nxt(), *mr_b2 = nxt();
  // pair_to_msa
  const float *pm_g = nxt(), *pm_bl = nxt(), *pm_W = nxt(), *pm_b = nxt();
  // msa_col
  const float *mc_g = nxt(), *mc_bl = nxt(), *mc_W1 = nxt(), *mc_b1 = nxt(),
              *mc_W2 = nxt(), *mc_b2 = nxt();
  // msa_trans
  const float *mt_g = nxt(), *mt_bl = nxt(), *mt_W1 = nxt(), *mt_b1 = nxt(),
              *mt_W2 = nxt(), *mt_b2 = nxt();
  // proj_a / proj_b
  const float *pa_g = nxt(), *pa_bl = nxt(), *pa_W = nxt(), *pa_b = nxt();
  const float *pbj_g = nxt(), *pbj_bl = nxt(), *pbj_W = nxt(), *pbj_b = nxt();
  // outer_to_pair
  const float *o2p_W = nxt(), *o2p_b = nxt();
  // tri_out, tri_in, attn_start, attn_end
  const float *to_g = nxt(), *to_bl = nxt(), *to_W1 = nxt(), *to_b1 = nxt(),
              *to_W2 = nxt(), *to_b2 = nxt();
  const float *ti_g = nxt(), *ti_bl = nxt(), *ti_W1 = nxt(), *ti_b1 = nxt(),
              *ti_W2 = nxt(), *ti_b2 = nxt();
  const float *as_g = nxt(), *as_bl = nxt(), *as_W1 = nxt(), *as_b1 = nxt(),
              *as_W2 = nxt(), *as_b2 = nxt();
  const float *ae_g = nxt(), *ae_bl = nxt(), *ae_W1 = nxt(), *ae_b1 = nxt(),
              *ae_W2 = nxt(), *ae_b2 = nxt();
  // pair_trans
  const float *pt_g = nxt(), *pt_bl = nxt(), *pt_W1 = nxt(), *pt_b1 = nxt(),
              *pt_W2 = nxt(), *pt_b2 = nxt();
  // time
  const float *tg_W1 = nxt(), *tg_b1 = nxt(), *tg_W2 = nxt(), *tg_b2 = nxt();

  // ---- workspace arena ----
  char* base = (char*)d_ws;
  size_t cur = 0;
  auto alloc = [&](size_t bytes) {
    void* r = base + cur;
    cur = (cur + bytes + 255) & ~(size_t)255;
    return r;
  };
  float* mix = (float*)alloc(8);
  bf16* wr1 = (bf16*)alloc(256 * 256 * 2);   // msa_row W1
  bf16* wr2 = (bf16*)alloc(256 * 256 * 2);
  bf16* wpm = (bf16*)alloc(128 * 256 * 2);   // pair_to_msa W
  bf16* wc1 = (bf16*)alloc(256 * 256 * 2);
  bf16* wc2 = (bf16*)alloc(256 * 256 * 2);
  bf16* wt1 = (bf16*)alloc(256 * 1024 * 2);
  bf16* wt2 = (bf16*)alloc(1024 * 256 * 2);
  bf16* wpa = (bf16*)alloc(256 * 64 * 2);    // proj_a W padded 32->64
  bf16* wpb = (bf16*)alloc(256 * 64 * 2);
  bf16* wop = (bf16*)alloc(1024 * 128 * 2);  // outer_to_pair W
  bf16* wo1 = (bf16*)alloc(128 * 128 * 2);   // tri_out
  bf16* wo2 = (bf16*)alloc(128 * 128 * 2);
  bf16* wi1 = (bf16*)alloc(128 * 128 * 2);   // tri_in
  bf16* wi2 = (bf16*)alloc(128 * 128 * 2);
  bf16* ws1 = (bf16*)alloc(128 * 128 * 2);   // attn_start
  bf16* ws2 = (bf16*)alloc(128 * 128 * 2);
  bf16* we1 = (bf16*)alloc(128 * 128 * 2);   // attn_end
  bf16* we2 = (bf16*)alloc(128 * 128 * 2);
  bf16* wp1 = (bf16*)alloc(128 * 512 * 2);   // pair_trans
  bf16* wp2 = (bf16*)alloc(512 * 128 * 2);

  bf16*  lnb    = (bf16*)alloc((size_t)SR * 256 * 2);    // 16MB LN output
  char*  shared = (char*)alloc((size_t)64 * 1024 * 1024);// hidden bf16 / pb f32
  float* mrow   = (float*)alloc((size_t)SR * 256 * 4);   // 32MB, later reused
  float* mci    = (float*)alloc((size_t)SR * 256 * 4);   // -> z ping
  float* mti    = (float*)alloc((size_t)SR * 256 * 4);   // -> z pong
  float* mout   = (float*)alloc((size_t)SR * 256 * 4);
  float* pbm    = (float*)alloc((size_t)256 * 256 * 4);

  bf16*  hb = (bf16*)shared;
  float* pb = (float*)shared;
  // mrow region reused after the MSA track:
  float* aproj = mrow;
  float* bproj = (float*)((char*)mrow + 4194304);
  bf16*  at    = (bf16*)((char*)mrow + 8388608);
  bf16*  bt    = (bf16*)((char*)mrow + 10485760);
  bf16*  opch  = (bf16*)((char*)mrow + 12582912);  // 16MB chunk
  float* zping = mci;
  float* zpong = mti;

  // ---- launch helpers ----
  auto conv = [&](const float* s, bf16* d, long n) {
    convert_bf16_kernel<<<dim3((unsigned)((n + 255) / 256)), 256, 0, stream>>>(s, d, n);
  };
  auto ln = [&](const float* X, const float* g, const float* b, bf16* Y,
                int rows, int d) {
    ln_bf16_kernel<<<dim3((unsigned)((rows + 7) / 8)), 256, 0, stream>>>(X, g, b, Y, rows, d);
  };
  auto gemm = [&](const bf16* A, const bf16* B, const float* bias,
                  const float* res, float* Cf, bf16* Cb, int M, int N, int K,
                  int Nout, int relu) {
    gemm_bf16_kernel<<<dim3(N / 64, M / 128), 256, 0, stream>>>(
        A, B, bias, res, Cf, Cb, M, N, K, Nout, relu);
  };

  // 1) time gate
  time_gate_kernel<<<1, 32, 0, stream>>>(t, tg_W1, tg_b1, tg_W2, tg_b2, mix);

  // 2) weight conversions f32 -> bf16
  conv(mr_W1, wr1, 256 * 256); conv(mr_W2, wr2, 256 * 256);
  conv(pm_W, wpm, 128 * 256);
  conv(mc_W1, wc1, 256 * 256); conv(mc_W2, wc2, 256 * 256);
  conv(mt_W1, wt1, 256 * 1024); conv(mt_W2, wt2, 1024 * 256);
  pad_weight_kernel<<<(256 * 64 + 255) / 256, 256, 0, stream>>>(pa_W, wpa, 256, 32, 64);
  pad_weight_kernel<<<(256 * 64 + 255) / 256, 256, 0, stream>>>(pbj_W, wpb, 256, 32, 64);
  conv(o2p_W, wop, 1024 * 128);
  conv(to_W1, wo1, 128 * 128); conv(to_W2, wo2, 128 * 128);
  conv(ti_W1, wi1, 128 * 128); conv(ti_W2, wi2, 128 * 128);
  conv(as_W1, ws1, 128 * 128); conv(as_W2, ws2, 128 * 128);
  conv(ae_W1, we1, 128 * 128); conv(ae_W2, we2, 128 * 128);
  conv(pt_W1, wp1, 128 * 512); conv(pt_W2, wp2, 512 * 128);

  // 3) MSA row block2: m_row = W2(relu(W1(LN(m))))
  ln(m, mr_g, mr_bl, lnb, SR, CM);
  gemm(lnb, wr1, mr_b1, nullptr, nullptr, hb, SR, 256, 256, 256, 1);
  gemm(hb, wr2, mr_b2, nullptr, mrow, nullptr, SR, 256, 256, 256, 0);

  // 4) pair_to_msa bias + mean over j
  ln(z, pm_g, pm_bl, lnb, RR, CZ);
  gemm(lnb, wpm, pm_b, nullptr, pb, nullptr, RR, 256, 128, 256, 0);
  mean_j_kernel<<<256, 256, 0, stream>>>(pb, pbm);

  // 5) m_col_in = m + m_row + pair_bias_mean
  mci_kernel<<<(SR * 256 + 255) / 256, 256, 0, stream>>>(m, mrow, pbm, mci,
                                                         (long)SR * 256);

  // 6) MSA col block2 (swapaxes is a no-op): m_trans_in = m_col_in + block2
  ln(mci, mc_g, mc_bl, lnb, SR, CM);
  gemm(lnb, wc1, mc_b1, nullptr, nullptr, hb, SR, 256, 256, 256, 1);
  gemm(hb, wc2, mc_b2, mci, mti, nullptr, SR, 256, 256, 256, 0);

  // 7) MSA transition (256 -> 1024 -> 256): m_out = m_trans_in + block2
  ln(mti, mt_g, mt_bl, lnb, SR, CM);
  gemm(lnb, wt1, mt_b1, nullptr, nullptr, hb, SR, 1024, 256, 1024, 1);
  gemm(hb, wt2, mt_b2, mti, mout, nullptr, SR, 256, 1024, 256, 0);

  // 8) projections a,b = LN(m_out) @ W (N padded 64, stored compact 32)
  ln(mout, pa_g, pa_bl, lnb, SR, CM);
  gemm(lnb, wpa, pa_b, nullptr, aproj, nullptr, SR, 64, 256, 32, 0);
  ln(mout, pbj_g, pbj_bl, lnb, SR, CM);
  gemm(lnb, wpb, pbj_b, nullptr, bproj, nullptr, SR, 64, 256, 32, 0);
  transpose_a_kernel<<<(256 * 32 * 128 + 255) / 256, 256, 0, stream>>>(aproj, at);
  conv(bproj, bt, (long)SR * 32);

  // 9) outer product mean + outer_to_pair, chunked over i; z1 = z + update
  for (int i0 = 0; i0 < 256; i0 += 32) {
    outer_kernel<<<dim3(32, 32), 256, 0, stream>>>(at, bt, opch, i0);
    gemm(opch, wop, o2p_b, z + (size_t)i0 * 256 * 128,
         zping + (size_t)i0 * 256 * 128, nullptr, 32 * 256, 128, 1024, 128, 0);
  }

  // 10) pair track: 4x (128->128->128) + transition (128->512->128)
  auto pairblk = [&](const float* zin, float* zout, const float* g,
                     const float* bl, const bf16* W1, const float* b1,
                     const bf16* W2, const float* b2, int hid) {
    ln(zin, g, bl, lnb, RR, CZ);
    gemm(lnb, W1, b1, nullptr, nullptr, hb, RR, hid, 128, hid, 1);
    gemm(hb, W2, b2, zin, zout, nullptr, RR, 128, hid, 128, 0);
  };
  pairblk(zping, zpong, to_g, to_bl, wo1, to_b1, wo2, to_b2, 128);  // tri_out
  pairblk(zpong, zping, ti_g, ti_bl, wi1, ti_b1, wi2, ti_b2, 128);  // tri_in
  pairblk(zping, zpong, as_g, as_bl, ws1, as_b1, ws2, as_b2, 128);  // attn_start
  pairblk(zpong, zping, ae_g, ae_bl, we1, ae_b1, we2, ae_b2, 128);  // attn_end (swap no-op)
  pairblk(zping, zpong, pt_g, pt_bl, wp1, pt_b1, wp2, pt_b2, 512);  // pair_trans

  // 11) outputs: dm_dt then dz_dt, concatenated
  float* out = (float*)d_out;
  final_kernel<<<(SR * 256 + 255) / 256, 256, 0, stream>>>(mout, m, mix, 0, out,
                                                           (long)SR * 256);
  final_kernel<<<(RR * 128 + 255) / 256, 256, 0, stream>>>(
      zpong, z, mix, 1, out + (size_t)SR * 256, (long)RR * 128);
}
